// GATLayer_16346645528742
// MI455X (gfx1250) — matrix operand order repaired and checked by reading, not compiled, not run on hardware
//
#include <hip/hip_runtime.h>
#include <hip/hip_bf16.h>

typedef __attribute__((ext_vector_type(16))) _Float16 v16h;
typedef __attribute__((ext_vector_type(8)))  float    v8f;

#define C_N      4096
#define IN_D     256
#define HD       256   // H*D
#define H_HEADS  4
#define D_DIM    64
#define NEG_SLOPE 0.2f
#define BIG_NEG  (-1.0e30f)
#define LOG2E    1.4426950408889634f

// bare v_exp_f32 (base-2 exponential) — hardware transcendental
__device__ __forceinline__ float fast_exp2(float x) {
  return __builtin_amdgcn_exp2f(x);
}

// ---------- fragment loaders (f32 memory -> f16 WMMA fragments) ----------
// A-fragment (16x32, M x K): lane&15 = M row. K per element:
//   lanes 0-15:  elems 0..7 -> K k0+0..7,   elems 8..15 -> K k0+16..23
//   lanes 16-31: elems 0..7 -> K k0+8..15,  elems 8..15 -> K k0+24..31
__device__ __forceinline__ v16h load_afrag_f32(const float* __restrict__ rowp,
                                               int k0, int koff) {
  const float4* p0 = (const float4*)(rowp + k0 + koff);
  const float4* p1 = (const float4*)(rowp + k0 + 16 + koff);
  float4 f0 = p0[0], f1 = p0[1], f2 = p1[0], f3 = p1[1];
  v16h a;
  a[0]=(_Float16)f0.x;  a[1]=(_Float16)f0.y;  a[2]=(_Float16)f0.z;  a[3]=(_Float16)f0.w;
  a[4]=(_Float16)f1.x;  a[5]=(_Float16)f1.y;  a[6]=(_Float16)f1.z;  a[7]=(_Float16)f1.w;
  a[8]=(_Float16)f2.x;  a[9]=(_Float16)f2.y;  a[10]=(_Float16)f2.z; a[11]=(_Float16)f2.w;
  a[12]=(_Float16)f3.x; a[13]=(_Float16)f3.y; a[14]=(_Float16)f3.z; a[15]=(_Float16)f3.w;
  return a;
}
// B-fragment (32x16, K x N): lane&15 = N col; K contiguous:
//   lanes 0-15: K k0+0..15 ; lanes 16-31: K k0+16..31
__device__ __forceinline__ v16h load_bfrag_f32(const float* __restrict__ rowp,
                                               int k0, int kb) {
  const float4* p = (const float4*)(rowp + k0 + kb);
  float4 f0 = p[0], f1 = p[1], f2 = p[2], f3 = p[3];
  v16h b;
  b[0]=(_Float16)f0.x;  b[1]=(_Float16)f0.y;  b[2]=(_Float16)f0.z;  b[3]=(_Float16)f0.w;
  b[4]=(_Float16)f1.x;  b[5]=(_Float16)f1.y;  b[6]=(_Float16)f1.z;  b[7]=(_Float16)f1.w;
  b[8]=(_Float16)f2.x;  b[9]=(_Float16)f2.y;  b[10]=(_Float16)f2.z; b[11]=(_Float16)f2.w;
  b[12]=(_Float16)f3.x; b[13]=(_Float16)f3.y; b[14]=(_Float16)f3.z; b[15]=(_Float16)f3.w;
  return b;
}

// ---------- kernel 1: h = x @ W.T via WMMA, f32 out ----------
__global__ void __launch_bounds__(128) proj_gemm_kernel(
    const float* __restrict__ x, const float* __restrict__ W,
    float* __restrict__ hfeat) {
  const int lane = threadIdx.x & 31;
  const int wave = threadIdx.x >> 5;
  const int tile = blockIdx.x * 4 + wave;       // 4096 tiles: 256 M x 16 N
  const int c0 = (tile & 255) * 16;
  const int n0 = (tile >> 8) * 16;
  const int koff = (lane & 16) ? 8 : 0;
  const int kb   = (lane & 16) ? 16 : 0;
  const float* xrow = x + (size_t)(c0 + (lane & 15)) * IN_D;
  const float* wrow = W + (size_t)(n0 + (lane & 15)) * IN_D;  // B[k][n]=W[n][k]
  v8f acc = {};
#pragma unroll
  for (int k0 = 0; k0 < IN_D; k0 += 32) {
    v16h a = load_bfrag_f32(xrow, k0, kb);
    v16h b = load_bfrag_f32(wrow, k0, kb);
    acc = __builtin_amdgcn_wmma_f32_16x16x32_f16(false, a, false, b,
                                                 (short)0, acc, false, false);
  }
  const int mbase = c0 + ((lane & 16) ? 8 : 0);
  const int n = lane & 15;
#pragma unroll
  for (int rr = 0; rr < 8; ++rr)
    hfeat[(size_t)(mbase + rr) * HD + n0 + n] = acc[rr];
}

// ---------- kernel 2: pack adjacency into a wave32-ballot bitmask ----------
__global__ void __launch_bounds__(256) pack_adj_kernel(
    const int* __restrict__ adj, unsigned* __restrict__ bits) {
  const size_t idx = (size_t)blockIdx.x * 256 + threadIdx.x;
  const int v = adj[idx];
  const unsigned m = (unsigned)__ballot(v != 0);   // wave32: 32 valid bits
  if ((threadIdx.x & 31) == 0) bits[idx >> 5] = m;
}

// ---------- kernel 3: attention logits, PRE-SCALED by log2(e) ----------
// lrelu is positively homogeneous, so softmax(lrelu(aS+aD)) can be computed
// base-2 on log2e-scaled logits: every exp becomes a single v_exp_f32.
__global__ void __launch_bounds__(256) alpha_kernel(
    const float* __restrict__ hfeat, const float* __restrict__ a_src,
    const float* __restrict__ a_dst, float* __restrict__ alS,
    float* __restrict__ alD) {
  const int c = blockIdx.x * blockDim.x + threadIdx.x;
  if (c >= C_N) return;
  const float* hrow = hfeat + (size_t)c * HD;
#pragma unroll
  for (int h = 0; h < H_HEADS; ++h) {
    float s = 0.f, d = 0.f;
#pragma unroll 8
    for (int k = 0; k < D_DIM; ++k) {
      float v = hrow[h * D_DIM + k];
      s = fmaf(v, a_src[h * D_DIM + k], s);
      d = fmaf(v, a_dst[h * D_DIM + k], d);
    }
    alS[h * C_N + c] = s * LOG2E;
    alD[h * C_N + c] = d * LOG2E;
  }
}

// ---------- kernel 4: V operand transpose + f16 convert: hT[hd][c] ----------
__global__ void __launch_bounds__(256) transpose_f16_kernel(
    const float* __restrict__ hfeat, _Float16* __restrict__ hT) {
  const int c  = blockIdx.x * 256 + threadIdx.x;
  const int hd = blockIdx.y;
  hT[(size_t)hd * C_N + c] = (_Float16)hfeat[(size_t)c * HD + hd];
}

// ---------- e generator for one 32-wide j chunk (contiguous K order) ------
// w: adjacency bits for [j0, j0+32); adstL: LDS-resident scaled alpha_dst
__device__ __forceinline__ void compute_e16(float e[16], unsigned w,
    const float* adstL, float as, int j0, int koff) {
  const float4* d0p = (const float4*)(adstL + j0 + koff);
  const float4* d1p = (const float4*)(adstL + j0 + 8 + koff);
  float4 d0 = d0p[0], d1 = d0p[1], d2 = d1p[0], d3 = d1p[1];
  const float dv[16] = {d0.x,d0.y,d0.z,d0.w, d1.x,d1.y,d1.z,d1.w,
                        d2.x,d2.y,d2.z,d2.w, d3.x,d3.y,d3.z,d3.w};
#pragma unroll
  for (int t = 0; t < 16; ++t) {
    const int k = koff + t;  // K in [0,32)
    float ee = as + dv[t];
    ee = (ee >= 0.f) ? ee : (NEG_SLOPE * ee);
    e[t] = ((w >> k) & 1u) ? ee : BIG_NEG;
  }
}

// ---------- kernel 5: fused flash-style masked softmax + P@V WMMA ----------
__global__ void __launch_bounds__(128) gat_attn_kernel(
    const unsigned* __restrict__ adjbits, const _Float16* __restrict__ hT,
    const float* __restrict__ alS, const float* __restrict__ alD,
    float* __restrict__ out) {
  __shared__ float s_adst[C_N];                 // 16 KB: scaled alpha_dst[head]

  const int lane = threadIdx.x & 31;
  const int wave = threadIdx.x >> 5;
  const int tile = blockIdx.x * 4 + wave;       // 1024 = 4 heads * 256 iblocks
  const int hh = blockIdx.x >> 6;               // head, uniform per block
  const int i0 = (tile & 255) * 16;
  const int row  = lane & 15;
  const int koff = (lane & 16) ? 8 : 0;         // unused; A-frag also splits by kb
  const int kb   = (lane & 16) ? 16 : 0;        // B-fragment K split

  // cooperative stage of alpha_dst[head] into LDS (all 4 waves share head)
  {
    const float4* src = (const float4*)(alD + (size_t)hh * C_N);
    float4* dst = (float4*)s_adst;
    for (int i = threadIdx.x; i < C_N / 4; i += 128) dst[i] = src[i];
  }
  __syncthreads();

  const unsigned* wrow = adjbits + (size_t)(i0 + row) * (C_N / 32);
  const float as = alS[(size_t)hh * C_N + i0 + row];

  // ---- pass 1: online row max + base-2 exp-sum (scalar state only) ----
  float m = BIG_NEG, l = 0.f;
  for (int j0 = 0; j0 < C_N; j0 += 32) {
    const unsigned w = wrow[j0 >> 5];
    float e[16];
    compute_e16(e, w, s_adst, as, j0, kb);
    float mloc = e[0];
#pragma unroll
    for (int t = 1; t < 16; ++t) mloc = fmaxf(mloc, e[t]);
    mloc = fmaxf(mloc, __shfl_xor(mloc, 16, 32));   // lane pair shares a row
    float mnew = fmaxf(m, mloc);
    float s = 0.f;
#pragma unroll
    for (int t = 0; t < 16; ++t) s += fast_exp2(e[t] - mnew); // bare v_exp_f32
    s += __shfl_xor(s, 16, 32);
    l = l * fast_exp2(m - mnew) + s;
    m = mnew;
  }
  const float invl = 1.0f / l;

  // ---- pass 2: P (f16 A-frag) x V (f16 B-frag) -> f32 accum via WMMA ----
  v8f acc0 = {}, acc1 = {}, acc2 = {}, acc3 = {};
  const _Float16* b0 = hT + (size_t)(hh * D_DIM +  0 + (lane & 15)) * C_N;
  const _Float16* b1 = b0 + (size_t)16 * C_N;
  const _Float16* b2 = b0 + (size_t)32 * C_N;
  const _Float16* b3 = b0 + (size_t)48 * C_N;

  for (int j0 = 0; j0 < C_N; j0 += 32) {
    if (j0 + 1024 < C_N) __builtin_prefetch(b0 + j0 + 1024, 0, 1);
    const unsigned w = wrow[j0 >> 5];
    float e[16];
    compute_e16(e, w, s_adst, as, j0, kb);
    v16h pa;                                     // unnormalized P row
#pragma unroll
    for (int t = 0; t < 16; ++t) pa[t] = (_Float16)fast_exp2(e[t] - m);
    v16h bb0 = *(const v16h*)(b0 + j0 + kb);     // 32B contiguous per lane
    v16h bb1 = *(const v16h*)(b1 + j0 + kb);
    v16h bb2 = *(const v16h*)(b2 + j0 + kb);
    v16h bb3 = *(const v16h*)(b3 + j0 + kb);
    acc0 = __builtin_amdgcn_wmma_f32_16x16x32_f16(false, pa, false, bb0, (short)0, acc0, false, false);
    acc1 = __builtin_amdgcn_wmma_f32_16x16x32_f16(false, pa, false, bb1, (short)0, acc1, false, false);
    acc2 = __builtin_amdgcn_wmma_f32_16x16x32_f16(false, pa, false, bb2, (short)0, acc2, false, false);
    acc3 = __builtin_amdgcn_wmma_f32_16x16x32_f16(false, pa, false, bb3, (short)0, acc3, false, false);
  }

  // ---- fold 1/l per row into C fragments (8 bpermute broadcasts) ----
  const int rbase = (lane & 16) ? 8 : 0;        // VGPR rr holds row rr+rbase
  const int mbase = i0 + rbase;
  const int n = lane & 15;
#pragma unroll
  for (int rr = 0; rr < 8; ++rr) {
    const float sc = __shfl(invl, rr + rbase, 32);   // lane r holds row r's 1/l
    float* orow = out + (size_t)(mbase + rr) * HD + hh * D_DIM;
    orow[n]      = acc0[rr] * sc;
    orow[16 + n] = acc1[rr] * sc;
    orow[32 + n] = acc2[rr] * sc;
    orow[48 + n] = acc3[rr] * sc;
  }
}

extern "C" void kernel_launch(void* const* d_in, const int* in_sizes, int n_in,
                              void* d_out, int out_size, void* d_ws, size_t ws_size,
                              hipStream_t stream) {
  const float* x     = (const float*)d_in[0];   // (4096, 256)
  const int*   adj   = (const int*)  d_in[1];   // (4096, 4096)
  const float* W     = (const float*)d_in[2];   // (256, 256)
  const float* a_src = (const float*)d_in[3];   // (4, 64)
  const float* a_dst = (const float*)d_in[4];   // (4, 64)
  float*       out   = (float*)d_out;           // (4096, 256)

  char* ws = (char*)d_ws;
  float*    hfeat   = (float*)ws;                             // 4 MB
  _Float16* hT      = (_Float16*)(ws + 4u * 1024 * 1024);     // 2 MB
  float*    alS     = (float*)(ws + 6u * 1024 * 1024);        // 64 KB
  float*    alD     = alS + H_HEADS * C_N;                    // 64 KB
  unsigned* adjbits = (unsigned*)(ws + 6u * 1024 * 1024 + 512u * 1024); // 2 MB

  proj_gemm_kernel<<<1024, 128, 0, stream>>>(x, W, hfeat);
  pack_adj_kernel<<<(C_N * C_N) / 256, 256, 0, stream>>>(adj, adjbits);
  alpha_kernel<<<C_N / 256, 256, 0, stream>>>(hfeat, a_src, a_dst, alS, alD);
  transpose_f16_kernel<<<dim3(C_N / 256, HD), 256, 0, stream>>>(hfeat, hT);
  gat_attn_kernel<<<256, 128, 0, stream>>>(adjbits, hT, alS, alD, out);
}
